// GRU_64381559767520
// MI455X (gfx1250) — matrix operand order repaired and checked by reading, not compiled, not run on hardware
//
#include <hip/hip_runtime.h>

// ---------------------------------------------------------------------------
// Persistent-wave GRU for MI455X (gfx1250, wave32).
//   grid  = 64 blocks x 32 threads (1 wave) ; each wave owns 16 batch rows
//   per step: 3 GEMMs [16,128]x[128,64] as 48x v_wmma_f32_16x16x32_f16
//   weights held in 384 VGPRs; h state in 32 VGPRs (4x v8f C-fragments)
//   embedding gather software-pipelined one step ahead through LDS
// ---------------------------------------------------------------------------

typedef __attribute__((ext_vector_type(16))) _Float16 v16h;
typedef __attribute__((ext_vector_type(8)))  _Float16 v8h;
typedef __attribute__((ext_vector_type(8)))  float    v8f;

#define LSEQ  512
#define BATCH 1024
#define EDIM  64
#define HDIM  64
#define ROWS  16      // batch rows per wave
#define STR   72      // LDS row stride in halves (144B -> 36-bank rotation, conflict-free frag loads)

static __device__ __forceinline__ v8f wmma_f16(v16h a, v16h b, v8f c) {
    // D = A(16x32 f16) x B(32x16 f16) + C(16x16 f32)
    return __builtin_amdgcn_wmma_f32_16x16x32_f16(false, a, false, b, (short)0, c, false, false);
}

static __device__ __forceinline__ float fast_sigmoid(float x) {
    return 1.0f / (1.0f + __expf(-x));
}
static __device__ __forceinline__ float fast_tanh(float x) {
    return 1.0f - 2.0f / (1.0f + __expf(2.0f * x));
}

// A-fragment (16x32 f16) loader from LDS row-major region with stride STR.
// Layout per ISA 7.12.2: lanes 0-15 hold rows m, elements 0-7 = K c0+0..7,
// elements 8-15 = K c0+16..23 ; lanes 16-31 get the +8 K-offsets.
static __device__ __forceinline__ v16h loadA(const _Float16* base, int c0, int m, int hi) {
    const int4* p0 = (const int4*)(base + m * STR + c0 + 8 * hi);
    const int4* p1 = (const int4*)(base + m * STR + c0 + 16 + 8 * hi);
    union { v16h v; int4 q[2]; } u;
    u.q[0] = *p0;
    u.q[1] = *p1;
    return u.v;
}

__global__ __launch_bounds__(32, 1)
void gru_persistent(const int* __restrict__ x, const float* __restrict__ emb,
                    const float* __restrict__ Wz, const float* __restrict__ Wr,
                    const float* __restrict__ Wh, float* __restrict__ out) {
    __shared__ int xidx[LSEQ * ROWS];                           // 32 KB: all indices for this tile
    __shared__ __align__(16) _Float16 xt2[2][ROWS * STR];       // double-buffered x_t (f16)
    __shared__ __align__(16) _Float16 hbuf [ROWS * STR];        // h_{t-1} as f16 rows
    __shared__ __align__(16) _Float16 rhbuf[ROWS * STR];        // r*h as f16 rows

    const int lane = threadIdx.x;           // 0..31
    const int b0   = blockIdx.x * ROWS;
    const int m    = lane & 15;
    const int hi   = lane >> 4;
    const int grow = lane >> 1;             // gather: row handled by this lane
    const int ghalf= lane & 1;              // gather: which 32-element half

    // ---- preload this tile's sequence indices (coalesced) ----
    for (int i = lane; i < LSEQ * ROWS; i += 32)
        xidx[i] = x[(i >> 4) * BATCH + b0 + (i & 15)];

    // ---- load weight B-fragments into registers -------------------------
    // B frag (kf,nf): lane holds column n = nf*16 + m,
    //                 rows k = kf*32 + hi*8 + e (e = 0..7), kf*32 + 16 + hi*8 + (e-8) (e = 8..15)
    // Wz/Wr: rows 0-63 multiply x_t, rows 64-127 multiply h  (matches [xt|h])
    // Wh   : rows 0-63 multiply r*h, rows 64-127 multiply x_t (matches [rh|xt])
    v16h bz[4][4], br[4][4], bh[4][4];
    #pragma unroll
    for (int kf = 0; kf < 4; ++kf) {
        #pragma unroll
        for (int nf = 0; nf < 4; ++nf) {
            const int kb = kf * 32 + hi * 8;
            #pragma unroll
            for (int e = 0; e < 16; ++e) {
                const int off = (kb + e + (e & 8)) * HDIM + nf * 16 + m;
                bz[kf][nf][e] = (_Float16)Wz[off];
                br[kf][nf][e] = (_Float16)Wr[off];
                bh[kf][nf][e] = (_Float16)Wh[off];
            }
        }
    }

    // ---- h = 0 ----
    v8f h[4];
    #pragma unroll
    for (int nf = 0; nf < 4; ++nf) h[nf] = (v8f){};

    // ---- prime pipeline: gather x_0 into xt2[0] ----
    {
        const int idx = xidx[grow];
        const float4* src = (const float4*)(emb + (size_t)idx * EDIM + ghalf * 32);
        float4 v[8];
        #pragma unroll
        for (int j = 0; j < 8; ++j) v[j] = src[j];
        _Float16* dst = &xt2[0][grow * STR + ghalf * 32];
        #pragma unroll
        for (int j = 0; j < 4; ++j) {
            union { v8h hv; int4 q; } u;
            const float* f = (const float*)&v[2 * j];
            #pragma unroll
            for (int i = 0; i < 8; ++i) u.hv[i] = (_Float16)f[i];
            ((int4*)dst)[j] = u.q;
        }
    }
    __syncthreads();

    // =========================== time loop ===============================
    for (int t = 0; t < LSEQ; ++t) {
        const int cur = t & 1;

        // -- issue next step's embedding-row loads early (hide latency) --
        const int tn = (t + 1 < LSEQ) ? (t + 1) : t;
        const int pidx = xidx[tn * ROWS + grow];
        const float4* psrc = (const float4*)(emb + (size_t)pidx * EDIM + ghalf * 32);
        float4 pv[8];
        #pragma unroll
        for (int j = 0; j < 8; ++j) pv[j] = psrc[j];

        // -- stage h (f16) to LDS: C-layout -> row-major --
        #pragma unroll
        for (int nf = 0; nf < 4; ++nf)
            #pragma unroll
            for (int g = 0; g < 8; ++g)
                hbuf[(g + 8 * hi) * STR + nf * 16 + m] = (_Float16)h[nf][g];
        __syncthreads();

        // -- A fragments for [x_t | h] (K = 128) --
        v16h a[4];
        a[0] = loadA(&xt2[cur][0], 0,  m, hi);
        a[1] = loadA(&xt2[cur][0], 32, m, hi);
        a[2] = loadA(hbuf,         0,  m, hi);
        a[3] = loadA(hbuf,         32, m, hi);

        // -- z and r gates: 32 WMMAs, 8 independent accumulator chains --
        v8f zacc[4], racc[4];
        #pragma unroll
        for (int nf = 0; nf < 4; ++nf) { zacc[nf] = (v8f){}; racc[nf] = (v8f){}; }
        #pragma unroll
        for (int nf = 0; nf < 4; ++nf) {
            #pragma unroll
            for (int kf = 0; kf < 4; ++kf) {
                zacc[nf] = wmma_f16(a[kf], bz[kf][nf], zacc[nf]);
                racc[nf] = wmma_f16(a[kf], br[kf][nf], racc[nf]);
            }
        }

        // -- sigmoids; stage r*h (f16) to LDS --
        #pragma unroll
        for (int nf = 0; nf < 4; ++nf) {
            #pragma unroll
            for (int g = 0; g < 8; ++g) {
                const float zz = fast_sigmoid(zacc[nf][g]);
                const float rr = fast_sigmoid(racc[nf][g]);
                zacc[nf][g] = zz;
                rhbuf[(g + 8 * hi) * STR + nf * 16 + m] = (_Float16)(rr * h[nf][g]);
            }
        }
        __syncthreads();

        // -- hcand = tanh([r*h | x_t] @ Wh): 16 WMMAs --
        const v16h c0 = loadA(rhbuf, 0,  m, hi);
        const v16h c1 = loadA(rhbuf, 32, m, hi);
        v8f cacc[4];
        #pragma unroll
        for (int nf = 0; nf < 4; ++nf) {
            v8f acc = (v8f){};
            acc = wmma_f16(c0,   bh[0][nf], acc);
            acc = wmma_f16(c1,   bh[1][nf], acc);
            acc = wmma_f16(a[0], bh[2][nf], acc);   // x_t frags reused from registers
            acc = wmma_f16(a[1], bh[3][nf], acc);
            cacc[nf] = acc;
        }

        // -- h update + streamed output store (half-wave 64B coalesced) --
        const size_t obase = ((size_t)t * BATCH + b0) * HDIM;
        #pragma unroll
        for (int nf = 0; nf < 4; ++nf) {
            #pragma unroll
            for (int g = 0; g < 8; ++g) {
                const float hc = fast_tanh(cacc[nf][g]);
                const float zz = zacc[nf][g];
                const float hv = (1.0f - zz) * h[nf][g] + zz * hc;
                h[nf][g] = hv;
                out[obase + (size_t)(g + 8 * hi) * HDIM + nf * 16 + m] = hv;
            }
        }

        // -- retire prefetch: convert + store into the other xt buffer --
        {
            _Float16* dst = &xt2[cur ^ 1][grow * STR + ghalf * 32];
            #pragma unroll
            for (int j = 0; j < 4; ++j) {
                union { v8h hv; int4 q; } u;
                const float* f = (const float*)&pv[2 * j];
                #pragma unroll
                for (int i = 0; i < 8; ++i) u.hv[i] = (_Float16)f[i];
                ((int4*)dst)[j] = u.q;
            }
        }
        // next iteration's __syncthreads orders these LDS writes vs. reads
    }

    // ---- h_last (second output, concatenated after outputs) ----
    const size_t hbase = (size_t)LSEQ * BATCH * HDIM + (size_t)b0 * HDIM;
    #pragma unroll
    for (int nf = 0; nf < 4; ++nf)
        #pragma unroll
        for (int g = 0; g < 8; ++g)
            out[hbase + (size_t)(g + 8 * hi) * HDIM + nf * 16 + m] = h[nf][g];
}

extern "C" void kernel_launch(void* const* d_in, const int* in_sizes, int n_in,
                              void* d_out, int out_size, void* d_ws, size_t ws_size,
                              hipStream_t stream) {
    const int*   x   = (const int*)  d_in[0];
    const float* emb = (const float*)d_in[1];
    const float* Wz  = (const float*)d_in[2];
    const float* Wr  = (const float*)d_in[3];
    const float* Wh  = (const float*)d_in[4];
    float* out = (float*)d_out;
    (void)in_sizes; (void)n_in; (void)out_size; (void)d_ws; (void)ws_size;

    gru_persistent<<<dim3(BATCH / ROWS), dim3(32), 0, stream>>>(x, emb, Wz, Wr, Wh, out);
}
